// AGNewsAE_46772193853592
// MI455X (gfx1250) — matrix operand order repaired and checked by reading, not compiled, not run on hardware
//
#include <hip/hip_runtime.h>
#include <hip/hip_bf16.h>
#include <math.h>

// ---------------- problem constants ----------------
#define T_SEQ   1024
#define HID     256
#define VOCAB   50000
#define SOS_TOK 2

// ---------------- launch shape ----------------
#define NWG        128            // persistent workgroups
#define NTHREADS   256            // 8 wave32 waves per WG
#define WAVES_WG   8
#define GWAVES     (NWG * WAVES_WG)   // 1024 waves total
#define NGRU       8              // WGs that own the recurrent GRU update

#define NTILES     ((VOCAB + 15) / 16)   // 3125 row-tiles of 16
#define KCHUNKS    (HID / 32)            // 8 K-chunks of 32 for bf16 WMMA

typedef __attribute__((ext_vector_type(16))) __bf16 v16bf;
typedef __attribute__((ext_vector_type(8)))  float  v8f;

// ---------------- workspace layout ----------------
struct WS {
  unsigned gridCnt, gridGen;           // full-grid barrier
  unsigned gruCnt,  gruGen;            // 8-WG GRU barrier
  int      h_ready;                    // last decoder step whose h is published
  int      tok;                        // greedy feedback token
  unsigned pad0, pad1;
  unsigned long long msPacked[T_SEQ];  // online softmax (m,s) packed per step
  unsigned long long argKey[T_SEQ];    // (monotonic logit, ~row) packed per step
  float    logZ[T_SEQ];
  float    h[HID];                     // fp32 recurrent hidden state
  unsigned short hbf[HID];             // bf16 copy for WMMA B operand
};
#define WSWZ_OFFSET 32768            // bf16-swizzled w_out starts here
static_assert(sizeof(WS) <= WSWZ_OFFSET, "WS too big");

struct SH {
  float eS[HID];
  float hS[HID];
  float part[6][32];
  __bf16 hbfS[HID];
  float redM[NTHREADS];
  float redS[NTHREADS];
  unsigned long long redK[NTHREADS];
  int   tokS;
};

// ---------------- helpers ----------------
__device__ __forceinline__ unsigned monoF(float f) {
  unsigned u = __float_as_uint(f);
  return (u & 0x80000000u) ? ~u : (u | 0x80000000u);
}

// WGP-scope prefetch: scope 0 pulls into ALL cache levels (incl. WGP$),
// unlike __builtin_prefetch's SE-scope lowering which stops at GL2.
__device__ __forceinline__ void prefetch_wgp(const void* p) {
  asm volatile("global_prefetch_b8 %0, off" :: "v"((unsigned long long)p));
}

// full-grid barrier; releasing thread finalizes step t (tok + logZ)
__device__ void gridSync(WS* ws, int t) {
  __syncthreads();
  if (threadIdx.x == 0) {
    __threadfence();
    unsigned g = atomicAdd(&ws->gridGen, 0u);
    if (atomicAdd(&ws->gridCnt, 1u) == NWG - 1) {
      if (t >= 0) {
        unsigned long long ms = atomicAdd(&ws->msPacked[t], 0ull);
        float m = __uint_as_float((unsigned)(ms & 0xFFFFFFFFull));
        float s = __uint_as_float((unsigned)(ms >> 32));
        ws->logZ[t] = m + logf(s);
        unsigned long long ak = atomicAdd(&ws->argKey[t], 0ull);
        ws->tok = (int)(0xFFFFFFFFu - (unsigned)(ak & 0xFFFFFFFFull));
      }
      __threadfence();
      atomicExch(&ws->gridCnt, 0u);
      __threadfence();
      atomicAdd(&ws->gridGen, 1u);
    } else {
      while (atomicAdd(&ws->gridGen, 0u) == g) __builtin_amdgcn_s_sleep(1);
      __threadfence();
    }
  }
  __syncthreads();
}

// barrier among the NGRU recurrent workgroups
__device__ void miniBarrier(WS* ws) {
  __syncthreads();
  if (threadIdx.x == 0) {
    __threadfence();
    unsigned g = atomicAdd(&ws->gruGen, 0u);
    if (atomicAdd(&ws->gruCnt, 1u) == NGRU - 1) {
      __threadfence();
      atomicExch(&ws->gruCnt, 0u);
      __threadfence();
      atomicAdd(&ws->gruGen, 1u);
    } else {
      while (atomicAdd(&ws->gruGen, 0u) == g) __builtin_amdgcn_s_sleep(1);
      __threadfence();
    }
  }
  __syncthreads();
}

// one GRU step for hidden slice [wg*32, wg*32+32); eS/hS already in LDS
__device__ void gru_compute(SH& sh, WS* ws, int wg,
                            const float* __restrict__ w_ih,
                            const float* __restrict__ w_hh,
                            const float* __restrict__ b_ih,
                            const float* __restrict__ b_hh) {
  int tid = threadIdx.x;
  int jBase = wg * 32;
  if (tid < 192) {
    int which = tid >> 5;                 // 0..5: {ih,hh} x {r,z,n}
    int jj    = tid & 31;
    int g     = which % 3;
    int mat   = which / 3;
    int rowIdx = g * HID + jBase + jj;
    const float* Wrow = (mat ? w_hh : w_ih) + (size_t)rowIdx * HID;
    const float* vec  = mat ? sh.hS : sh.eS;
    float acc = (mat ? b_hh : b_ih)[rowIdx];
#pragma unroll 8
    for (int k = 0; k < HID; ++k) acc = fmaf(Wrow[k], vec[k], acc);
    sh.part[which][jj] = acc;
  }
  __syncthreads();
  if (tid < 32) {
    int j = jBase + tid;
    float ir = sh.part[0][tid], iz = sh.part[1][tid], in_ = sh.part[2][tid];
    float hr = sh.part[3][tid], hz = sh.part[4][tid], hn  = sh.part[5][tid];
    float r = 1.f / (1.f + __expf(-(ir + hr)));
    float z = 1.f / (1.f + __expf(-(iz + hz)));
    float n = tanhf(in_ + r * hn);
    float hnew = (1.f - z) * n + z * sh.hS[j];
    ws->h[j] = hnew;
    __bf16 hb = (__bf16)hnew;
    ws->hbf[j] = *(unsigned short*)&hb;
  }
  __syncthreads();
}

// ---------------- init kernel (runs before persistent kernel) ----------------
__global__ void ws_reset_kernel(WS* ws) {
  ws->gridCnt = 0u; ws->gridGen = 0u;
  ws->gruCnt  = 0u; ws->gruGen  = 0u;
  ws->h_ready = -1;
  ws->tok     = SOS_TOK;
}

// ---------------- persistent seq2seq kernel ----------------
__global__ __launch_bounds__(NTHREADS)
void seq2seq_persistent(const int* __restrict__ x,
                        const float* __restrict__ enc_emb,
                        const float* __restrict__ enc_w_ih, const float* __restrict__ enc_w_hh,
                        const float* __restrict__ enc_b_ih, const float* __restrict__ enc_b_hh,
                        const float* __restrict__ dec_emb,
                        const float* __restrict__ dec_w_ih, const float* __restrict__ dec_w_hh,
                        const float* __restrict__ dec_b_ih, const float* __restrict__ dec_b_hh,
                        const float* __restrict__ w_out, const float* __restrict__ b_out,
                        float* __restrict__ out, WS* ws, __bf16* wswz) {
  __shared__ SH sh;
  const int tid   = threadIdx.x;
  const int wg    = blockIdx.x;
  const int wave  = tid >> 5;
  const int lane  = tid & 31;
  const int half  = lane >> 4;          // A/B fragment half-wave selector
  const int gwave = wg * WAVES_WG + wave;

  // ---- Phase 0: swizzle w_out -> bf16 in WMMA-A fragment order ----
  // A 16x32 bf16 layout: lane = {half, M=lane&15}; element j -> K = half*8 + (j<8 ? j : j+8)
  for (int tile = gwave; tile < NTILES; tile += GWAVES) {
    int row = tile * 16 + (lane & 15);
#pragma unroll
    for (int c = 0; c < KCHUNKS; ++c) {
      __bf16* dst = wswz + ((((size_t)tile * KCHUNKS + c) * 32 + lane) * 16);
      const float* src = w_out + (size_t)row * HID + c * 32;
#pragma unroll
      for (int j = 0; j < 16; ++j) {
        int K = half * 8 + ((j < 8) ? j : j + 8);
        dst[j] = (__bf16)src[K];
      }
    }
  }
  int gtid = wg * NTHREADS + tid;
  unsigned long long negPack =
      ((unsigned long long)__float_as_uint(0.0f) << 32) |
      (unsigned long long)__float_as_uint(-__builtin_inff());
  for (int t = gtid; t < T_SEQ; t += NWG * NTHREADS) {
    ws->msPacked[t] = negPack;
    ws->argKey[t]   = 0ull;
  }
  if (gtid < HID) ws->h[gtid] = 0.0f;     // encoder initial hidden
  __threadfence();
  gridSync(ws, -1);

  // ---- Phase 1: encoder (sequential GRU scan; 8 WGs partition hidden dim) ----
  if (wg < NGRU) {
    for (int t = 0; t < T_SEQ; ++t) {
      int token = x[t];
      sh.hS[tid] = ((volatile float*)ws->h)[tid];
      sh.eS[tid] = enc_emb[(size_t)token * HID + tid];
      miniBarrier(ws);   // all GRU WGs captured h before anyone overwrites it
      gru_compute(sh, ws, wg, enc_w_ih, enc_w_hh, enc_b_ih, enc_b_hh);
      __threadfence();
      miniBarrier(ws);   // all slices of new h visible
    }
  }

  // ---- Phase 2: decoder (GRU + WMMA GEMV + online log-softmax + argmax) ----
  for (int t = 0; t < T_SEQ; ++t) {
    if (wg < NGRU) {
      if (tid == 0) sh.tokS = ((volatile int*)&ws->tok)[0];
      __syncthreads();
      int token = sh.tokS;
      sh.hS[tid] = ((volatile float*)ws->h)[tid];
      float ev = dec_emb[(size_t)token * HID + tid];
      sh.eS[tid] = ev > 0.f ? ev : 0.f;   // embed + relu
      miniBarrier(ws);
      gru_compute(sh, ws, wg, dec_w_ih, dec_w_hh, dec_b_ih, dec_b_hh);
      __threadfence();
      miniBarrier(ws);
      if (wg == 0 && tid == 0) atomicExch(&ws->h_ready, t);
    }
    if (tid == 0) {
      while (atomicAdd(&ws->h_ready, 0) < t) __builtin_amdgcn_s_sleep(1);
    }
    __syncthreads();
    {
      unsigned short hv = ((volatile unsigned short*)ws->hbf)[tid];
      sh.hbfS[tid] = *(__bf16*)&hv;
    }
    __syncthreads();

    // B fragment: 32x16 bf16, lanes 0-15 hold K=0..7,16..23, lanes 16-31 K=8..15,24..31;
    // h replicated across all 16 N columns (GEMV), so every column of D = W*h.
    v16bf bfrag[KCHUNKS];
#pragma unroll
    for (int c = 0; c < KCHUNKS; ++c) {
#pragma unroll
      for (int j = 0; j < 16; ++j)
        bfrag[c][j] = sh.hbfS[c * 32 + half * 8 + ((j < 8) ? j : j + 8)];
    }

    float m = -__builtin_inff(), s = 0.f;
    unsigned long long best = 0ull;
    for (int tile = gwave; tile < NTILES; tile += GWAVES) {
      // Prefetch the next tile's 8KB fragment block into the WGP cache while
      // this tile computes: 32 lanes x 128B x 2 covers the whole region.
      int nxt = tile + GWAVES;
      if (nxt < NTILES) {
        const char* p = (const char*)(wswz + (size_t)nxt * (KCHUNKS * 512));
        prefetch_wgp(p + lane * 128);
        prefetch_wgp(p + 4096 + lane * 128);
      }

      v8f acc = {0.f, 0.f, 0.f, 0.f, 0.f, 0.f, 0.f, 0.f};
      const __bf16* ap = wswz + (((size_t)tile * KCHUNKS) * 32 + lane) * 16;
#pragma unroll
      for (int c = 0; c < KCHUNKS; ++c) {
        v16bf a = *(const v16bf*)(ap + (size_t)c * 512);
        acc = __builtin_amdgcn_wmma_f32_16x16x32_bf16(
            false, a, false, bfrag[c], (short)0, acc, false, false);
      }
      // D column 0: lanes 0/16 hold rows tile*16 + half*8 + r in acc[r]
      if ((lane & 15) == 0) {
        int rbase = tile * 16 + half * 8;
#pragma unroll
        for (int r = 0; r < 8; ++r) {
          int row = rbase + r;
          float logit = acc[r] + b_out[row];
          out[(size_t)t * VOCAB + row] = logit;   // raw logit; normalized later
          float nm = fmaxf(m, logit);
          s = s * __expf(m - nm) + __expf(logit - nm);
          m = nm;
          unsigned long long key =
              ((unsigned long long)monoF(logit) << 32) |
              (unsigned long long)(0xFFFFFFFFu - (unsigned)row);
          best = best > key ? best : key;
        }
      }
    }

    // block-level online-softmax / argmax reduction
    sh.redM[tid] = m; sh.redS[tid] = s; sh.redK[tid] = best;
    __syncthreads();
    if (tid == 0) {
      float bm = -__builtin_inff(), bs = 0.f;
      unsigned long long bk = 0ull;
      for (int i = 0; i < NTHREADS; ++i) {
        float om = sh.redM[i];
        if (om != -__builtin_inff()) {
          float os = sh.redS[i];
          float nm = fmaxf(bm, om);
          bs = bs * __expf(bm - nm) + os * __expf(om - nm);
          bm = nm;
        }
        if (sh.redK[i] > bk) bk = sh.redK[i];
      }
      atomicMax(&ws->argKey[t], bk);
      unsigned long long old = atomicAdd(&ws->msPacked[t], 0ull);
      for (;;) {
        float gm = __uint_as_float((unsigned)(old & 0xFFFFFFFFull));
        float gs = __uint_as_float((unsigned)(old >> 32));
        float nm = fmaxf(gm, bm);
        float ns = gs * __expf(gm - nm) + bs * __expf(bm - nm);
        unsigned long long neu =
            ((unsigned long long)__float_as_uint(ns) << 32) |
            (unsigned long long)__float_as_uint(nm);
        unsigned long long prev = atomicCAS(&ws->msPacked[t], old, neu);
        if (prev == old) break;
        old = prev;
      }
    }
    __threadfence();
    gridSync(ws, t);   // releaser computes tok + logZ[t] for next step
  }

  // ---- Phase 3: normalize logits -> log-probs ----
  for (int t = wg; t < T_SEQ; t += NWG) {
    float lz = ((volatile float*)ws->logZ)[t];
    float* o = out + (size_t)t * VOCAB;
    for (int v = tid; v < VOCAB; v += NTHREADS) o[v] -= lz;
  }
}

// ---------------- host entry ----------------
extern "C" void kernel_launch(void* const* d_in, const int* in_sizes, int n_in,
                              void* d_out, int out_size, void* d_ws, size_t ws_size,
                              hipStream_t stream) {
  (void)in_sizes; (void)n_in; (void)out_size; (void)ws_size;
  const int*   x        = (const int*)d_in[0];
  const float* enc_emb  = (const float*)d_in[1];
  const float* enc_w_ih = (const float*)d_in[2];
  const float* enc_w_hh = (const float*)d_in[3];
  const float* enc_b_ih = (const float*)d_in[4];
  const float* enc_b_hh = (const float*)d_in[5];
  const float* dec_emb  = (const float*)d_in[6];
  const float* dec_w_ih = (const float*)d_in[7];
  const float* dec_w_hh = (const float*)d_in[8];
  const float* dec_b_ih = (const float*)d_in[9];
  const float* dec_b_hh = (const float*)d_in[10];
  const float* w_out    = (const float*)d_in[11];
  const float* b_out    = (const float*)d_in[12];

  WS*     ws   = (WS*)d_ws;
  __bf16* wswz = (__bf16*)((char*)d_ws + WSWZ_OFFSET);  // needs ~25.7 MB of ws

  ws_reset_kernel<<<1, 1, 0, stream>>>(ws);
  seq2seq_persistent<<<NWG, NTHREADS, 0, stream>>>(
      x, enc_emb, enc_w_ih, enc_w_hh, enc_b_ih, enc_b_hh,
      dec_emb, dec_w_ih, dec_w_hh, dec_b_ih, dec_b_hh,
      w_out, b_out, (float*)d_out, ws, wswz);
}
